// Navi_Diego_alt_69827578298543
// MI455X (gfx1250) — compile-verified, matrix operand order checked by reading, not run on hardware
//
#include <hip/hip_runtime.h>
#include <hip/hip_bf16.h>

// ---------------------------------------------------------------------------
// RGCN layer for MI455X (gfx1250, wave32, WMMA).
//   out = diag(1/full_deg) * sum_rel [ diag(1/deg_r) * (A_r X + deg_r * b_r) ] W_r
// - scatter: HBM-bound edge gather + hardware f32 atomics; edge lists NT so
//   L2 (192 MB) keeps X (102 MB, gathered 16x) and agg (102 MB RMW) resident
// - GEMM: v_wmma_f32_16x16x32_bf16, f32 accumulate, 32x256 block tile.
//   A-staging fuses bias+degree-normalize AND zero-clears agg for the next
//   relation. B tile staged via global_load_async_to_lds_b128 (ASYNCcnt) from
//   a pre-transposed bf16 W, so one B fragment = two ds_load_b128, zero cvts.
//   C-writeback is nontemporal (out is touched once per relation).
// ---------------------------------------------------------------------------

typedef __bf16        v16bf __attribute__((ext_vector_type(16)));
typedef float         v8f   __attribute__((ext_vector_type(8)));
typedef unsigned int  v8u   __attribute__((ext_vector_type(8)));

#define D_DIM 256
#define TILE_M 32
#define TILE_K 32
#define BS_STRIDE 20   // uints per column row in LDS B tile (16 data + 4 pad)

// ---------------- zero fill (float4 stores) ----------------
__global__ __launch_bounds__(256) void zero4_kernel(float4* __restrict__ p, long n4) {
    long i = (long)blockIdx.x * blockDim.x + threadIdx.x;
    if (i < n4) p[i] = make_float4(0.f, 0.f, 0.f, 0.f);
}

// ---------------- W -> transposed bf16: WT[col][k] -------------------------
__global__ __launch_bounds__(256) void wprep_kernel(const float* __restrict__ W,
                                                    __bf16* __restrict__ WT) {
    int idx = blockIdx.x * 256 + threadIdx.x;   // 65536 elements
    int c = idx >> 8, k = idx & 255;
    WT[(size_t)c * D_DIM + k] = (__bf16)W[(size_t)k * D_DIM + c];
}

// ---------------- edge scatter: agg[row] += X[col]; deg[row] += 1 ----------
__global__ __launch_bounds__(256) void scatter_kernel(const float* __restrict__ X,
                                                      const int*  __restrict__ rows,
                                                      const int*  __restrict__ cols,
                                                      float* __restrict__ agg,
                                                      float* __restrict__ deg,
                                                      int nedges) {
    int tid  = threadIdx.x;
    int edge = blockIdx.x * 4 + (tid >> 6);
    if (edge >= nedges) return;
    int l = tid & 63;
    int r = __builtin_nontemporal_load(&rows[edge]);   // streamed once: keep out of L2
    int c = __builtin_nontemporal_load(&cols[edge]);
    const float4 x = ((const float4*)(X + (size_t)c * D_DIM))[l];
    float* dst = agg + (size_t)r * D_DIM + l * 4;
    unsafeAtomicAdd(dst + 0, x.x);
    unsafeAtomicAdd(dst + 1, x.y);
    unsafeAtomicAdd(dst + 2, x.z);
    unsafeAtomicAdd(dst + 3, x.w);
    if (l == 0) unsafeAtomicAdd(deg + r, 1.0f);
}

// ---------------- fused normalize + WMMA GEMM ------------------------------
// out += [diag(1/deg)*(agg + deg*bias)](bf16) @ W(bf16);  agg cleared to 0.
// Block: 256 threads = 8 waves, tile 32 rows x 256 cols. Wave w owns a 32x32
// C slab: 2 A-frags x 2 B-frags -> 4 v_wmma per 32-wide K chunk.
__global__ __launch_bounds__(256) void gemm_acc_kernel(float* __restrict__ Y,
                                                       const __bf16* __restrict__ WT,
                                                       const float* __restrict__ bias,
                                                       const float* __restrict__ deg,
                                                       float* __restrict__ out,
                                                       int nrows) {
    __shared__ float        As[TILE_M][TILE_K + 4];   // 4.6 KB, 144 B row stride
    __shared__ unsigned int BsT[D_DIM][BS_STRIDE];    // 20 KB, bf16 pairs [col][k/2]

    const int tid  = threadIdx.x;
    const int lane = tid & 31;
    const int wid  = tid >> 5;                 // column group 0..7
    const int rowBase = blockIdx.x * TILE_M;

    v8f acc00 = {}, acc01 = {}, acc10 = {}, acc11 = {};

    // 16-bit WMMA operand layout (wave32):
    //   lanes 0-15 : halves 0..7 -> K 0..7,  halves 8..15 -> K 16..23
    //   lanes 16-31: halves 0..7 -> K 8..15, halves 8..15 -> K 24..31
    const int laneHalf = (lane & 16) ? 8 : 0;
    const int uOff  = laneHalf >> 1;           // uint (bf16-pair) offset: 0 or 4
    const int aRow0 = (lane & 15);
    const int aRow1 = aRow0 + 16;
    const int bCol0 = wid * 32 + (lane & 15);

    // A staging map: 32x32 floats = 256 float4, one per thread
    const int aR = tid >> 3, aC4 = (tid & 7) * 4;
    const float dg  = deg[rowBase + aR];       // deg padded to NPAD, zeroed
    const float inv = (dg == 0.f) ? 1.f : (1.f / dg);

    for (int k0 = 0; k0 < D_DIM; k0 += TILE_K) {
        // ---- stage B^T tile (bf16) with async copy to LDS (ASYNCcnt):
        //      256 cols x 16 uints -> 4 b128 transfers per thread, L2-resident
        #pragma unroll
        for (int i = 0; i < 4; ++i) {
            int lin = i * 256 + tid;           // uint4 index, 0..1023
            int c = lin >> 2, j4 = (lin & 3) * 4;
            unsigned lds_off = (unsigned)(size_t)&BsT[c][j4];
            unsigned long long ga = (unsigned long long)(size_t)
                ((const unsigned int*)(WT + (size_t)c * D_DIM + k0) + j4);
            asm volatile("global_load_async_to_lds_b128 %0, %1, off"
                         :: "v"(lds_off), "v"(ga)
                         : "memory");
        }

        // ---- stage A: load agg, fused bias+normalize, zero-back for next rel
        float* srcA = &Y[(size_t)(rowBase + aR) * D_DIM + k0 + aC4];
        float4 f  = *(const float4*)srcA;
        *(float4*)srcA = make_float4(0.f, 0.f, 0.f, 0.f);
        float4 b4 = *(const float4*)&bias[k0 + aC4];
        float4 y;
        y.x = inv * fmaf(dg, b4.x, f.x);
        y.y = inv * fmaf(dg, b4.y, f.y);
        y.z = inv * fmaf(dg, b4.z, f.z);
        y.w = inv * fmaf(dg, b4.w, f.w);
        *(float4*)&As[aR][aC4] = y;

        asm volatile("s_wait_asynccnt 0" ::: "memory");
        __syncthreads();

        // ---- A fragments: f32 -> bf16 in registers
        v16bf a0, a1;
        #pragma unroll
        for (int h = 0; h < 16; ++h) {
            int kk = laneHalf + ((h < 8) ? h : (h + 8));
            a0[h] = (__bf16)As[aRow0][kk];
            a1[h] = (__bf16)As[aRow1][kk];
        }
        // ---- B fragments: two ds_load_b128 each, no conversion
        uint4 l0 = *(const uint4*)&BsT[bCol0][uOff];
        uint4 h0 = *(const uint4*)&BsT[bCol0][uOff + 8];
        uint4 l1 = *(const uint4*)&BsT[bCol0 + 16][uOff];
        uint4 h1 = *(const uint4*)&BsT[bCol0 + 16][uOff + 8];
        v8u bu0 = {l0.x, l0.y, l0.z, l0.w, h0.x, h0.y, h0.z, h0.w};
        v8u bu1 = {l1.x, l1.y, l1.z, l1.w, h1.x, h1.y, h1.z, h1.w};
        v16bf b0 = __builtin_bit_cast(v16bf, bu0);
        v16bf b1 = __builtin_bit_cast(v16bf, bu1);

        acc00 = __builtin_amdgcn_wmma_f32_16x16x32_bf16(false, a0, false, b0,
                                                        (short)0, acc00, false, false);
        acc01 = __builtin_amdgcn_wmma_f32_16x16x32_bf16(false, a0, false, b1,
                                                        (short)0, acc01, false, false);
        acc10 = __builtin_amdgcn_wmma_f32_16x16x32_bf16(false, a1, false, b0,
                                                        (short)0, acc10, false, false);
        acc11 = __builtin_amdgcn_wmma_f32_16x16x32_bf16(false, a1, false, b1,
                                                        (short)0, acc11, false, false);
        __syncthreads();
    }

    // C layout: VGPR v -> row M = v (+8 for lanes 16-31), N = lane&15.
    // out is touched once per relation and re-read only a full pass later:
    // nontemporal RMW keeps it from evicting X/agg out of L2.
    const int mOff = (lane & 16) ? 8 : 0;
    const int cCol = wid * 32 + (lane & 15);
    if (rowBase + TILE_M <= nrows) {           // block-uniform fast path
        #pragma unroll
        for (int v = 0; v < 8; ++v) {
            size_t o0 = (size_t)(rowBase + v + mOff) * D_DIM + cCol;
            size_t o1 = o0 + (size_t)16 * D_DIM;
            __builtin_nontemporal_store(
                __builtin_nontemporal_load(&out[o0]) + acc00[v], &out[o0]);
            __builtin_nontemporal_store(
                __builtin_nontemporal_load(&out[o0 + 16]) + acc01[v], &out[o0 + 16]);
            __builtin_nontemporal_store(
                __builtin_nontemporal_load(&out[o1]) + acc10[v], &out[o1]);
            __builtin_nontemporal_store(
                __builtin_nontemporal_load(&out[o1 + 16]) + acc11[v], &out[o1 + 16]);
        }
    } else {
        #pragma unroll
        for (int v = 0; v < 8; ++v) {
            int gr0 = rowBase + v + mOff;
            int gr1 = gr0 + 16;
            if (gr0 < nrows) {
                size_t o = (size_t)gr0 * D_DIM + cCol;
                out[o]      += acc00[v];
                out[o + 16] += acc01[v];
            }
            if (gr1 < nrows) {
                size_t o = (size_t)gr1 * D_DIM + cCol;
                out[o]      += acc10[v];
                out[o + 16] += acc11[v];
            }
        }
    }
}

// ---------------- final normalization --------------------------------------
__global__ __launch_bounds__(256) void invfull_kernel(const float* __restrict__ deg_all,
                                                      float* __restrict__ inv_full,
                                                      int nrows, int npad) {
    int n = blockIdx.x * blockDim.x + threadIdx.x;
    if (n >= nrows) return;
    float s = 0.f;
    #pragma unroll
    for (int r = 0; r < 16; ++r)
        s += (deg_all[(size_t)r * npad + n] > 0.f) ? 1.f : 0.f;
    inv_full[n] = 1.f / ((s == 0.f) ? 1.f : s);
}

__global__ __launch_bounds__(256) void scale_kernel(float* __restrict__ out,
                                                    const float* __restrict__ inv_full) {
    int n = blockIdx.x;
    int d = threadIdx.x;
    out[(size_t)n * D_DIM + d] *= inv_full[n];
}

// ---------------------------------------------------------------------------
extern "C" void kernel_launch(void* const* d_in, const int* in_sizes, int n_in,
                              void* d_out, int out_size, void* d_ws, size_t ws_size,
                              hipStream_t stream) {
    const float* features = (const float*)d_in[0];
    const float* w        = (const float*)d_in[1];
    const float* bias     = (const float*)d_in[2];
    const float* w_t      = (const float*)d_in[3];
    const float* bias_t   = (const float*)d_in[4];
    const int* adj_row    = (const int*)d_in[5];
    const int* adj_col    = (const int*)d_in[6];
    const int* adj_t_row  = (const int*)d_in[7];
    const int* adj_t_col  = (const int*)d_in[8];
    float* out = (float*)d_out;

    const int N = in_sizes[0] / D_DIM;                     // 100000
    const int E = in_sizes[5] / 8;                         // 262144 per relation
    const int NPAD = ((N + TILE_M - 1) / TILE_M) * TILE_M; // row-padded for GEMM

    // workspace: deg_all[16*NPAD] | inv_full[N] | WT[256*256 bf16] | agg[NPAD*256]
    float*  deg_all  = (float*)d_ws;
    float*  inv_full = deg_all + (size_t)16 * NPAD;
    __bf16* wbf      = (__bf16*)(inv_full + N);
    float*  agg      = (float*)(wbf + (size_t)D_DIM * D_DIM);

    const long outN4 = (long)N * D_DIM / 4;
    const long aggN4 = (long)NPAD * D_DIM / 4;
    const long degN4 = ((long)16 * NPAD + (long)N) / 4;    // deg_all + inv_full

    zero4_kernel<<<(int)((outN4 + 255) / 256), 256, 0, stream>>>((float4*)out, outN4);
    zero4_kernel<<<(int)((degN4 + 255) / 256), 256, 0, stream>>>((float4*)deg_all, degN4);
    zero4_kernel<<<(int)((aggN4 + 255) / 256), 256, 0, stream>>>((float4*)agg, aggN4);

    const int gemmGrid = NPAD / TILE_M;

    for (int rel = 0; rel < 16; ++rel) {
        const int  r = rel & 7;
        const bool t = rel >= 8;
        const float* Wr   = (t ? w_t    : w)    + (size_t)r * D_DIM * D_DIM;
        const float* br   = (t ? bias_t : bias) + (size_t)r * D_DIM;
        const int*   rows = (t ? adj_t_row : adj_row) + (size_t)r * E;
        const int*   cols = (t ? adj_t_col : adj_col) + (size_t)r * E;
        float* degr = deg_all + (size_t)rel * NPAD;

        wprep_kernel<<<D_DIM * D_DIM / 256, 256, 0, stream>>>(Wr, wbf);
        scatter_kernel<<<(E + 3) / 4, 256, 0, stream>>>(features, rows, cols, agg, degr, E);
        // gemm also zero-clears agg, ready for the next relation's scatter
        gemm_acc_kernel<<<gemmGrid, 256, 0, stream>>>(agg, wbf, br, degr, out, N);
    }

    invfull_kernel<<<(N + 255) / 256, 256, 0, stream>>>(deg_all, inv_full, N, NPAD);
    scale_kernel<<<N, 256, 0, stream>>>(out, inv_full);
}
